// spatialTemporalLearningLayer_32023276159123
// MI455X (gfx1250) — matrix operand order repaired and checked by reading, not compiled, not run on hardware
//
#include <hip/hip_runtime.h>
#include <hip/hip_bf16.h>

// ---- types ------------------------------------------------------------
typedef __attribute__((ext_vector_type(16))) __bf16 v16bf;
typedef __attribute__((ext_vector_type(8)))  float  v8f;
typedef __attribute__((ext_vector_type(4)))  unsigned int v4u;
typedef __attribute__((ext_vector_type(8)))  int v8i;
typedef __attribute__((ext_vector_type(4)))  int v4i;

union FragBF { v16bf v; v4u q[2]; };   // 32 bytes = one WMMA bf16 operand

#define BT_TOT 48      // B*T = 4*12
#define NN 1024        // nodes
#define FF 64          // features
#define NEG_INF_F (-9.0e15f)

#ifndef __has_builtin
#define __has_builtin(x) 0
#endif
#if __has_builtin(__builtin_amdgcn_tensor_load_to_lds) && __has_builtin(__builtin_amdgcn_s_wait_tensorcnt)
#define HAVE_TDM 1
#else
#define HAVE_TDM 0
#endif

// ---- kernel 0: pack adjacency into bit matrix (1024 x 32 words) -------
__global__ void stl_pack_adj(const int* __restrict__ adj, unsigned* __restrict__ bits) {
  int w = blockIdx.x * blockDim.x + threadIdx.x;     // word index
  if (w >= NN * (NN / 32)) return;
  int row = w >> 5;
  int colbase = (w & 31) << 5;
  const int* p = adj + (size_t)row * NN + colbase;
  unsigned m = 0;
#pragma unroll
  for (int b = 0; b < 32; ++b) m |= (p[b] > 0) ? (1u << b) : 0u;
  bits[w] = m;
}

// ---- kernel 1: xf[bt][n][o] = sum_f x[bt][n][f] * theta[o][f]  (bf16 out)
__global__ __launch_bounds__(128) void stl_theta_proj(const float* __restrict__ x,
                                                      const float* __restrict__ theta,
                                                      __bf16* __restrict__ xf) {
  const int bt = blockIdx.x >> 4;        // 48 slices
  const int rtile = blockIdx.x & 15;     // 16 tiles of 64 rows
  const int wave = threadIdx.x >> 5;
  const int lane = threadIdx.x & 31;
  const int ln = lane & 15;
  const int hi = lane >> 4;
  const int rowA = rtile * 64 + wave * 16 + ln;
  const float* xr = x + ((size_t)bt * NN + rowA) * FF;

  v16bf a[2];
#pragma unroll
  for (int kw = 0; kw < 2; ++kw) {
#pragma unroll
    for (int e = 0; e < 8; ++e) {
      a[kw][e]     = (__bf16)xr[kw * 32 + hi * 16 + e];
      a[kw][8 + e] = (__bf16)xr[kw * 32 + hi * 16 + 8 + e];
    }
  }

#pragma unroll
  for (int ot = 0; ot < 4; ++ot) {
    const float* tr = theta + (size_t)(ot * 16 + ln) * FF;
    v16bf b0, b1;
#pragma unroll
    for (int e = 0; e < 16; ++e) {
      b0[e] = (__bf16)tr[hi * 16 + e];
      b1[e] = (__bf16)tr[32 + hi * 16 + e];
    }
    v8f c = {};
    c = __builtin_amdgcn_wmma_f32_16x16x32_bf16(false, a[0], false, b0, (short)0, c, false, false);
    c = __builtin_amdgcn_wmma_f32_16x16x32_bf16(false, a[1], false, b1, (short)0, c, false, false);
#pragma unroll
    for (int v = 0; v < 8; ++v) {
      int row = rtile * 64 + wave * 16 + v + 8 * hi;
      xf[((size_t)bt * NN + row) * FF + ot * 16 + ln] = (__bf16)c[v];
    }
  }
}

// ---- TDM helper: DMA one 32x64 bf16 tile from global into LDS ---------
__device__ inline unsigned lds_offset(const void* p) {
  return (unsigned)(unsigned long long)(size_t)p;   // low 32 bits of flat addr = LDS offset
}

#if HAVE_TDM
__device__ inline void tdm_load_block(unsigned lds_addr, const __bf16* gsrc) {
  unsigned long long ga = (unsigned long long)(size_t)gsrc;
  v4u g0;
  g0[0] = 1u;                                                  // count=1 (valid user D#)
  g0[1] = lds_addr;                                            // lds_addr [63:32]
  g0[2] = (unsigned)ga;                                        // global_addr [95:64]
  g0[3] = ((unsigned)(ga >> 32) & 0x01FFFFFFu) | (2u << 30);   // addr[56:32] | type=2
  v8i g1;
  g1[0] = 1 << 16;            // data_size = 1 -> 2 bytes (bf16)
  g1[1] = (int)(64u << 16);   // tensor_dim0 = 64 (lo16 at [63:48])
  g1[2] = (int)(1024u << 16); // tensor_dim1 = 1024 (lo16 at [95:80])
  g1[3] = (int)(64u << 16);   // tile_dim0 = 64  ([127:112])
  g1[4] = 32;                 // tile_dim1 = 32  ([143:128])
  g1[5] = 64;                 // tensor_dim0_stride = 64 ([207:160] lo32)
  g1[6] = 0;
  g1[7] = 0;
  v4i z4 = {0, 0, 0, 0};
  v8i z8 = {0, 0, 0, 0, 0, 0, 0, 0};
  // 6-arg toolchain form: (g0, g1, g2, g3, g4, cpol)
  __builtin_amdgcn_tensor_load_to_lds(g0, g1, z4, z4, z8, 0);
}
#endif

// ---- kernel 2: fused masked attention (flash-style, bf16 WMMA, fp32 acc)
// 4 waves/block, each wave owns 32 query rows (two 16-row A tiles) so K/V
// B-fragments are reused across two WMMA tiles. K/V blocks double-buffered
// in LDS, filled by the Tensor Data Mover (TENSORcnt) when available.
__global__ __launch_bounds__(128) void stl_attention(const __bf16* __restrict__ xf,
                                                     const unsigned* __restrict__ adjbits,
                                                     float* __restrict__ out) {
  __shared__ __bf16  kbuf[2][32][64];     // double-buffered K/V block (8 KB)
  __shared__ unsigned adjb[128][32];      // adjacency bits for 128 query rows (16 KB)
  __shared__ __bf16  pstage[4][16][32];   // per-wave P relayout buffer (4 KB)

  const int bt    = blockIdx.x >> 3;
  const int qbase = (blockIdx.x & 7) * 128;
  const int wave  = threadIdx.x >> 5;
  const int lane  = threadIdx.x & 31;
  const int ln    = lane & 15;
  const int hi    = lane >> 4;

  for (int i = threadIdx.x; i < 128 * 32; i += 128)
    adjb[i >> 5][i & 31] = adjbits[(size_t)(qbase + (i >> 5)) * 32 + (i & 31)];

  const __bf16* kv = xf + (size_t)bt * NN * FF;

  // Q fragments for both 16-row tiles of this wave (A layout), kept in regs
  v16bf qa[2][2];
#pragma unroll
  for (int qt = 0; qt < 2; ++qt) {
    const __bf16* qr = xf + ((size_t)bt * NN + qbase + wave * 32 + qt * 16 + ln) * FF;
#pragma unroll
    for (int kw = 0; kw < 2; ++kw) {
      FragBF f;
      f.q[0] = *(const v4u*)(qr + kw * 32 + hi * 16);
      f.q[1] = *(const v4u*)(qr + kw * 32 + hi * 16 + 8);
      qa[qt][kw] = f.v;
    }
  }

  v8f o[2][4] = {};
  float mrow[2][8], lrow[2][8];
#pragma unroll
  for (int qt = 0; qt < 2; ++qt)
#pragma unroll
    for (int v = 0; v < 8; ++v) { mrow[qt][v] = -3.0e38f; lrow[qt][v] = 0.0f; }

#if HAVE_TDM
  if (wave == 0) tdm_load_block(lds_offset(&kbuf[0][0][0]), kv);
#endif

  for (int j = 0; j < 32; ++j) {
    const int cur = j & 1;
#if HAVE_TDM
    if (wave == 0) __builtin_amdgcn_s_wait_tensorcnt(0);
    __syncthreads();                              // buf[cur] ready; buf[cur^1] free
    if (wave == 0 && j + 1 < 32)
      tdm_load_block(lds_offset(&kbuf[cur ^ 1][0][0]), kv + (size_t)(j + 1) * 32 * FF);
#else
    __syncthreads();
    {   // cooperative fallback copy: 32x64 bf16 = 4 KB, 32 B per thread
      int idx = threadIdx.x * 16;
      int r = idx >> 6, cs = idx & 63;
      const v4u* src = (const v4u*)(kv + (size_t)(j * 32 + r) * FF + cs);
      v4u* dst = (v4u*)(&kbuf[cur][r][cs]);
      dst[0] = src[0];
      dst[1] = src[1];
    }
    __syncthreads();
#endif

    // ---- S = Q @ K^T : B fragments shared by both query tiles ----
    v8f s[2][2];
#pragma unroll
    for (int t = 0; t < 2; ++t) {
      const __bf16* krow = &kbuf[cur][t * 16 + ln][0];
      FragBF b0, b1;
      b0.q[0] = *(const v4u*)(krow + hi * 16);
      b0.q[1] = *(const v4u*)(krow + hi * 16 + 8);
      b1.q[0] = *(const v4u*)(krow + 32 + hi * 16);
      b1.q[1] = *(const v4u*)(krow + 32 + hi * 16 + 8);
#pragma unroll
      for (int qt = 0; qt < 2; ++qt) {
        v8f c = {};
        c = __builtin_amdgcn_wmma_f32_16x16x32_bf16(false, qa[qt][0], false, b0.v, (short)0, c, false, false);
        c = __builtin_amdgcn_wmma_f32_16x16x32_bf16(false, qa[qt][1], false, b1.v, (short)0, c, false, false);
        s[qt][t] = c;
      }
    }

    // ---- mask + scale + online softmax ----
    float pr0[2][8], pr1[2][8], sfv[2][8];
#pragma unroll
    for (int qt = 0; qt < 2; ++qt) {
#pragma unroll
      for (int v = 0; v < 8; ++v) {
        unsigned w = adjb[wave * 32 + qt * 16 + v + 8 * hi][j];
        float sA = ((w >> ln) & 1u)        ? s[qt][0][v] * 0.125f : NEG_INF_F;
        float sB = ((w >> (ln + 16)) & 1u) ? s[qt][1][v] * 0.125f : NEG_INF_F;
        float bm = fmaxf(sA, sB);
#pragma unroll
        for (int off = 1; off < 16; off <<= 1)
          bm = fmaxf(bm, __shfl_xor(bm, off, 32));
        float mn = fmaxf(mrow[qt][v], bm);
        float scale = __expf(mrow[qt][v] - mn);
        float p0 = __expf(sA - mn);
        float p1 = __expf(sB - mn);
        float rs = p0 + p1;
#pragma unroll
        for (int off = 1; off < 16; off <<= 1)
          rs += __shfl_xor(rs, off, 32);
        lrow[qt][v] = lrow[qt][v] * scale + rs;
        mrow[qt][v] = mn;
        sfv[qt][v] = scale; pr0[qt][v] = p0; pr1[qt][v] = p1;
      }
    }

    // rescale running output
#pragma unroll
    for (int qt = 0; qt < 2; ++qt)
#pragma unroll
      for (int ot = 0; ot < 4; ++ot)
#pragma unroll
        for (int v = 0; v < 8; ++v) o[qt][ot][v] *= sfv[qt][v];

    // ---- V column gathers, shared across both query tiles ----
    v16bf bv[4];
#pragma unroll
    for (int ot = 0; ot < 4; ++ot)
#pragma unroll
      for (int e = 0; e < 16; ++e)
        bv[ot][e] = kbuf[cur][hi * 16 + e][ot * 16 + ln];

    // ---- O += P @ V per query tile (P relayout via per-wave LDS) ----
#pragma unroll
    for (int qt = 0; qt < 2; ++qt) {
#pragma unroll
      for (int v = 0; v < 8; ++v) {
        pstage[wave][v + 8 * hi][ln]      = (__bf16)pr0[qt][v];
        pstage[wave][v + 8 * hi][16 + ln] = (__bf16)pr1[qt][v];
      }
      FragBF pa;
      pa.q[0] = *(const v4u*)(&pstage[wave][ln][hi * 16]);
      pa.q[1] = *(const v4u*)(&pstage[wave][ln][hi * 16 + 8]);
#pragma unroll
      for (int ot = 0; ot < 4; ++ot)
        o[qt][ot] = __builtin_amdgcn_wmma_f32_16x16x32_bf16(false, pa.v, false, bv[ot], (short)0, o[qt][ot], false, false);
    }
  }

  // ---- epilogue: normalize, ReLU, store fp32 ----
#pragma unroll
  for (int qt = 0; qt < 2; ++qt) {
    float inv[8];
#pragma unroll
    for (int v = 0; v < 8; ++v) inv[v] = 1.0f / lrow[qt][v];
#pragma unroll
    for (int ot = 0; ot < 4; ++ot)
#pragma unroll
      for (int v = 0; v < 8; ++v) {
        int row = qbase + wave * 32 + qt * 16 + v + 8 * hi;
        out[((size_t)bt * NN + row) * FF + ot * 16 + ln] = fmaxf(o[qt][ot][v] * inv[v], 0.0f);
      }
  }
}

// ---- launcher ---------------------------------------------------------
extern "C" void kernel_launch(void* const* d_in, const int* in_sizes, int n_in,
                              void* d_out, int out_size, void* d_ws, size_t ws_size,
                              hipStream_t stream) {
  const float* x     = (const float*)d_in[0];   // (4,12,1024,64) fp32
  const int*   adj   = (const int*)d_in[1];     // (1024,1024) int32
  const float* theta = (const float*)d_in[2];   // (64,64) fp32
  float* out = (float*)d_out;                   // (4,12,1024,64) fp32

  __bf16* xf = (__bf16*)d_ws;                                       // 6.29 MB
  unsigned* adjbits = (unsigned*)((char*)d_ws +
                       (size_t)BT_TOT * NN * FF * sizeof(__bf16));  // +128 KB

  stl_pack_adj<<<(NN * (NN / 32) + 255) / 256, 256, 0, stream>>>(adj, adjbits);
  stl_theta_proj<<<BT_TOT * 16, 128, 0, stream>>>(x, theta, xf);
  stl_attention<<<BT_TOT * 8, 128, 0, stream>>>(xf, adjbits, out);
}
